// AirModel_20083267076939
// MI455X (gfx1250) — matrix operand order repaired and checked by reading, not compiled, not run on hardware
//
#include <hip/hip_runtime.h>
#include <hip/hip_bf16.h>

// Problem constants (from reference): B=512, T=2048, I=13, H=50
#define BB   512
#define TT   2048
#define IN   13
#define HID  50
#define HP   64      // padded hidden
#define GP   256     // padded 4*H gates
#define NTHREADS 128 // 4 waves of 32

typedef __attribute__((ext_vector_type(16))) __bf16 v16bf;
typedef __attribute__((ext_vector_type(8)))  __bf16 v8bf;
typedef __attribute__((ext_vector_type(8)))  float  v8f;

// ---- fast activations: CDNA5 hardware v_tanh_f32 ----
static __device__ __forceinline__ float fast_tanh(float x) {
#if __has_builtin(__builtin_amdgcn_tanhf)
    return __builtin_amdgcn_tanhf(x);
#else
    float e = __expf(2.0f * x);
    return (e - 1.0f) * __builtin_amdgcn_rcpf(e + 1.0f);
#endif
}
// sigmoid(x) = 0.5*tanh(0.5*x) + 0.5 ; caller passes pre-scaled arg = 0.5*x (+0.5*bias)
static __device__ __forceinline__ float sigm_from_half(float halfarg) {
    return __builtin_fmaf(0.5f, fast_tanh(halfarg), 0.5f);
}

static __device__ __forceinline__ v8f wmma_bf16(v16bf a, v16bf b, v8f c) {
    return __builtin_amdgcn_wmma_f32_16x16x32_bf16(
        /*neg_a=*/false, a, /*neg_b=*/false, b,
        /*c_mod=*/(short)0, c, /*reuse_a=*/false, /*reuse_b=*/false);
}

union U16 { v16bf v; v8bf h2[2]; __bf16 e[16]; };

__global__ __launch_bounds__(NTHREADS)
void lstm_fused_kernel(const float* __restrict__ x,
                       const int*   __restrict__ lengths,
                       const float* __restrict__ W_ih,
                       const float* __restrict__ W_hh,
                       const float* __restrict__ b_ih,
                       const float* __restrict__ b_hh,
                       const float* __restrict__ W_out,
                       const float* __restrict__ b_out,
                       float* __restrict__ out)
{
    // Padded weights (bf16) + bias (f32) + double-buffered h, all in LDS.
    __shared__ __align__(32) __bf16 sWhh[GP * HP];     // 32 KB  [256][64]
    __shared__ __align__(32) __bf16 sWih[GP * 32];     // 16 KB  [256][32] (K padded 13->32)
    __shared__ __align__(32) __bf16 sWout[16 * HP];    //  2 KB  [16][64]
    __shared__ float  sBias[GP];                       //  1 KB
    __shared__ __align__(32) __bf16 sH[2][16 * HP];    //  4 KB  ping-pong h(t)

    const int tid = threadIdx.x;
    const int w   = tid >> 5;          // wave id 0..3 -> owns hidden cols [16w,16w+16)
    const int l   = tid & 31;          // lane
    const int ln  = l & 15;
    const int hi  = l >> 4;            // lane half
    const int b0  = blockIdx.x * 16;   // batch tile base

    // ---- zero LDS pads ----
    for (int i = tid; i < GP * HP; i += NTHREADS) sWhh[i]  = (__bf16)0.f;
    for (int i = tid; i < GP * 32; i += NTHREADS) sWih[i]  = (__bf16)0.f;
    for (int i = tid; i < 16 * HP; i += NTHREADS) sWout[i] = (__bf16)0.f;
    for (int i = tid; i < GP;      i += NTHREADS) sBias[i] = 0.f;
    for (int i = tid; i < 2 * 16 * HP; i += NTHREADS) (&sH[0][0])[i] = (__bf16)0.f;
    __syncthreads();

    // ---- scatter real weights into padded LDS ----
    for (int idx = tid; idx < 4 * HID * HID; idx += NTHREADS) {       // W_hh [200][50]
        int r = idx / HID, k = idx % HID;
        int pr = (r / HID) * HP + (r % HID);
        sWhh[pr * HP + k] = (__bf16)W_hh[idx];
    }
    for (int idx = tid; idx < 4 * HID * IN; idx += NTHREADS) {        // W_ih [200][13]
        int r = idx / IN, k = idx % IN;
        int pr = (r / HID) * HP + (r % HID);
        sWih[pr * 32 + k] = (__bf16)W_ih[idx];
    }
    for (int idx = tid; idx < IN * HID; idx += NTHREADS) {            // W_out [13][50]
        int n = idx / HID, k = idx % HID;
        sWout[n * HP + k] = (__bf16)W_out[idx];
    }
    for (int idx = tid; idx < 4 * HID; idx += NTHREADS) {             // bias
        int pr = (idx / HID) * HP + (idx % HID);
        sBias[pr] = b_ih[idx] + b_hh[idx];
    }
    __syncthreads();

    // ---- persistent B fragments (B[k][n] = W[n][k]); lane ln = N col ----
    const int kofs = hi ? 8 : 0;
    v16bf Bih[4], Bhh[4][2], Bo0, Bo1;
#pragma unroll
    for (int g = 0; g < 4; ++g) {
        int nrow = (g * 4 + w) * 16 + ln;
        Bih[g]    = __builtin_shufflevector(*(const v8bf*)(sWih + nrow * 32 + kofs), *(const v8bf*)(sWih + nrow * 32 + kofs + 16), 0,1,2,3,4,5,6,7,8,9,10,11,12,13,14,15);
        Bhh[g][0] = __builtin_shufflevector(*(const v8bf*)(sWhh + nrow * HP + 0  + kofs), *(const v8bf*)(sWhh + nrow * HP + 0  + kofs + 16), 0,1,2,3,4,5,6,7,8,9,10,11,12,13,14,15);
        Bhh[g][1] = __builtin_shufflevector(*(const v8bf*)(sWhh + nrow * HP + 32 + kofs), *(const v8bf*)(sWhh + nrow * HP + 32 + kofs + 16), 0,1,2,3,4,5,6,7,8,9,10,11,12,13,14,15);
    }
    Bo0 = __builtin_shufflevector(*(const v8bf*)(sWout + ln * HP + 0  + kofs), *(const v8bf*)(sWout + ln * HP + 0  + kofs + 16), 0,1,2,3,4,5,6,7,8,9,10,11,12,13,14,15);
    Bo1 = __builtin_shufflevector(*(const v8bf*)(sWout + ln * HP + 32 + kofs), *(const v8bf*)(sWout + ln * HP + 32 + kofs + 16), 0,1,2,3,4,5,6,7,8,9,10,11,12,13,14,15);

    // half-bias for sigmoid gates (i,f,o) and full bias for g gate
    const float hb_i = 0.5f * sBias[(0 * 4 + w) * 16 + ln];
    const float hb_f = 0.5f * sBias[(1 * 4 + w) * 16 + ln];
    const float b_g  =        sBias[(2 * 4 + w) * 16 + ln];
    const float hb_o = 0.5f * sBias[(3 * 4 + w) * 16 + ln];
    const float bout_l = (ln < IN) ? b_out[ln] : 0.f;

    int lenv[8];
    long obase[8];                      // flat output offset at t=0 for each owned row
#pragma unroll
    for (int v = 0; v < 8; ++v) {
        int row = hi * 8 + v;
        lenv[v]  = lengths[b0 + row];
        obase[v] = (long)(b0 + row) * TT * IN + ln;
    }

    // per-lane f32 cell/hidden state; shared zero C vector (loop-invariant)
    v8f cst, hst, zvec;
#pragma unroll
    for (int v = 0; v < 8; ++v) { cst[v] = 0.f; hst[v] = 0.f; zvec[v] = 0.f; }

    const int hk0 = hi ? 8 : 0;        // A-frag K base within each 32-chunk

    // ================= main recurrence over time =================
    for (int t = 0; t < TT; ++t) {
        const int par = t & 1;
        const __bf16* hb = sH[par ^ 1]; // masked h(t-1)

        // ---- h(t-1) A-fragments (16-bit 16x32 layout) ----
        U16 ha0, ha1;
        ha0.h2[0] = *(const v8bf*)(hb + ln * HP + hk0);
        ha0.h2[1] = *(const v8bf*)(hb + ln * HP + hk0 + 16);
        ha1.h2[0] = *(const v8bf*)(hb + ln * HP + 32 + hk0);
        ha1.h2[1] = *(const v8bf*)(hb + ln * HP + 32 + hk0 + 16);

        // ---- wave 0: output projection for step t-1 (pipelined) ----
        if (w == 0 && t > 0) {
            v8f y = wmma_bf16(ha0.v, Bo0, zvec);
            y = wmma_bf16(ha1.v, Bo1, y);
            if (ln < IN) {
                const int tm1 = t - 1;
                const long oofs = obase[0] - ln + (long)tm1 * IN + ln; // == obase-style base
#pragma unroll
                for (int v = 0; v < 8; ++v) {
                    float val = bout_l + ((tm1 < lenv[v]) ? y[v] : 0.f);
                    __builtin_nontemporal_store(val, out + obase[v] + (long)tm1 * IN);
                }
                (void)oofs;
            }
        }

        // ---- x(t) A-fragment from global (K padded 13->32) ----
        U16 xa;
#pragma unroll
        for (int i = 0; i < 16; ++i) xa.e[i] = (__bf16)0.f;
        const float* xr = x + ((size_t)(b0 + ln) * TT + t) * IN;
        if (hi == 0) {
#pragma unroll
            for (int j = 0; j < 8; ++j) xa.e[j] = (__bf16)xr[j];
        } else {
#pragma unroll
            for (int j = 0; j < 5; ++j) xa.e[j] = (__bf16)xr[8 + j];
        }
        __builtin_prefetch(xr + IN, 0, 3);   // next timestep's x row

        // ---- gates = x*Wih^T + h*Whh^T  (bias folded into activations) ----
        v8f acc[4];
#pragma unroll
        for (int g = 0; g < 4; ++g) {
            v8f a = wmma_bf16(xa.v, Bih[g], zvec);
            a = wmma_bf16(ha0.v, Bhh[g][0], a);
            a = wmma_bf16(ha1.v, Bhh[g][1], a);
            acc[g] = a;
        }

        // ---- elementwise LSTM cell (f32), masking, publish h(t) ----
        __bf16* hs = sH[par];
#pragma unroll
        for (int v = 0; v < 8; ++v) {
            float ig = sigm_from_half(__builtin_fmaf(0.5f, acc[0][v], hb_i));
            float fg = sigm_from_half(__builtin_fmaf(0.5f, acc[1][v], hb_f));
            float gg = fast_tanh(acc[2][v] + b_g);
            float og = sigm_from_half(__builtin_fmaf(0.5f, acc[3][v], hb_o));
            float cn = __builtin_fmaf(fg, cst[v], ig * gg);
            float hn = og * fast_tanh(cn);
            bool valid = t < lenv[v];
            cst[v] = valid ? cn : cst[v];
            hst[v] = valid ? hn : hst[v];
            hs[(hi * 8 + v) * HP + (w * 16 + ln)] = (__bf16)hst[v];
        }
        __syncthreads();
    }

    // ---- drain: output for final step t = TT-1 ----
    if (w == 0) {
        const __bf16* hb = sH[(TT - 1) & 1];
        U16 ha0, ha1;
        ha0.h2[0] = *(const v8bf*)(hb + ln * HP + hk0);
        ha0.h2[1] = *(const v8bf*)(hb + ln * HP + hk0 + 16);
        ha1.h2[0] = *(const v8bf*)(hb + ln * HP + 32 + hk0);
        ha1.h2[1] = *(const v8bf*)(hb + ln * HP + 32 + hk0 + 16);
        v8f y = wmma_bf16(ha0.v, Bo0, zvec);
        y = wmma_bf16(ha1.v, Bo1, y);
        if (ln < IN) {
            const int tm1 = TT - 1;
#pragma unroll
            for (int v = 0; v < 8; ++v) {
                float val = bout_l + ((tm1 < lenv[v]) ? y[v] : 0.f);
                __builtin_nontemporal_store(val, out + obase[v] + (long)tm1 * IN);
            }
        }
    }
}

extern "C" void kernel_launch(void* const* d_in, const int* in_sizes, int n_in,
                              void* d_out, int out_size, void* d_ws, size_t ws_size,
                              hipStream_t stream) {
    const float* x       = (const float*)d_in[0];
    const int*   lengths = (const int*)  d_in[1];
    const float* W_ih    = (const float*)d_in[2];
    const float* W_hh    = (const float*)d_in[3];
    const float* b_ih    = (const float*)d_in[4];
    const float* b_hh    = (const float*)d_in[5];
    const float* W_out   = (const float*)d_in[6];
    const float* b_out   = (const float*)d_in[7];
    float* out = (float*)d_out;

    dim3 grid(BB / 16);        // 32 workgroups, one 16-row batch tile each
    dim3 block(NTHREADS);      // 4 waves of 32
    lstm_fused_kernel<<<grid, block, 0, stream>>>(x, lengths, W_ih, W_hh,
                                                  b_ih, b_hh, W_out, b_out, out);
}
